// PENGUIN_7782480740553
// MI455X (gfx1250) — matrix operand order repaired and checked by reading, not compiled, not run on hardware
//
#include <hip/hip_runtime.h>
#include <math.h>

#define BB 32
#define TT 4096
#define HH 64
#define PP 32
#define MHID 256
#define NBLK_ 4
#define NTOK (BB*TT)          /* 131072 tokens */
#define NH   ((size_t)NTOK*HH)/* 8388608 */

typedef __attribute__((ext_vector_type(16))) _Float16 v16h;
typedef __attribute__((ext_vector_type(8)))  float    v8f;

__device__ __forceinline__ v8f wmma32f16(v16h a, v16h b, v8f c) {
  // D = A(16x32,f16) * B(32x16,f16) + C(16x16,f32)
  return __builtin_amdgcn_wmma_f32_16x16x32_f16(false, a, false, b, (short)0, c, false, false);
}

// A-operand K index for element i of v16h (contiguous order, matches KMAPB)
#define KMAPA(i, hs) ((i) + ((hs) << 4))
// B-operand K index for element i (16-bit B 32x16)
#define KMAPB(i, hs) ((i) + ((hs) << 4))

// ---------------------------------------------------------------- elementwise / prep
__global__ void __launch_bounds__(256) k_xt(const float* __restrict__ tgt,
                                            const float* __restrict__ x0,
                                            const float* __restrict__ ts,
                                            float* __restrict__ xt) {
  size_t i = (size_t)blockIdx.x * 256 + threadIdx.x;
  int b = (int)(i / TT);
  float t = ts[b];
  xt[i] = (1.f - t) * x0[i] + t * tgt[i];
}

__global__ void __launch_bounds__(256) k_zero(float* __restrict__ a) {
  a[(size_t)blockIdx.x * 256 + threadIdx.x] = 0.f;
}

__global__ void __launch_bounds__(256) k_add(const float* __restrict__ a,
                                             const float* __restrict__ b,
                                             float* __restrict__ c) {
  size_t i = (size_t)blockIdx.x * 256 + threadIdx.x;
  c[i] = a[i] + b[i];
}

// float -> f16 weight pre-conversion (runs once per launch; weights are L2-resident)
__global__ void __launch_bounds__(256) k_cvt(const float* __restrict__ src,
                                             _Float16* __restrict__ dst) {
  size_t i = (size_t)blockIdx.x * 256 + threadIdx.x;
  dst[i] = (_Float16)src[i];
}

// Bbar[p,h] = ((Abar-1)/lam) * Bm[p,h]  as f16, all NBLK blocks at once: idx=(i,p,h)
__global__ void __launch_bounds__(256) k_bbar(const float* __restrict__ lam_log,
                                              const float* __restrict__ Bm,
                                              const float* __restrict__ log_dt,
                                              _Float16* __restrict__ Bh) {
  size_t idx = (size_t)blockIdx.x * 256 + threadIdx.x;
  int h = (int)(idx & 63);
  int pi = (int)((idx >> 6) & 31);
  int i = (int)(idx >> 11);
  float lam = -expf(lam_log[i * PP + pi]);
  float A = expf(lam * expf(log_dt[i * PP + pi]));
  float coef = (A - 1.f) / lam;
  Bh[idx] = (_Float16)(coef * Bm[(size_t)i * PP * HH + pi * 64 + h]);
}

// ---------------------------------------------------------------- conv1 (1->H) + SiLU, VALU (small)
__global__ void __launch_bounds__(256) k_conv1(const float* __restrict__ sig,
                                               const float* __restrict__ w,
                                               const float* __restrict__ bias,
                                               float* __restrict__ out) {
  size_t flat = (size_t)blockIdx.x * 256 + threadIdx.x;   // (b, h, t)
  int t = (int)(flat % TT);
  int h = (int)((flat / TT) % HH);
  int b = (int)(flat / ((size_t)TT * HH));
  float acc = bias[h];
#pragma unroll
  for (int k = 0; k < 32; ++k) {
    int ti = t + k - 15;
    if (ti >= 0 && ti < TT) acc += w[h * 32 + k] * sig[(size_t)b * TT + ti];
  }
  out[flat] = acc / (1.f + expf(-acc));                    // SiLU fused
}

// ---------------------------------------------------------------- conv2 (H->H) as implicit GEMM, WMMA, f16 weights
// in: (B, 64, T) channel-major;  out: (B, T, 64) token-major (fuses the transpose)
__global__ void __launch_bounds__(128) k_conv2(const float* __restrict__ in,
                                               const _Float16* __restrict__ w,  // (64, 64, 32) f16
                                               const float* __restrict__ bias,
                                               float* __restrict__ out) {
  __shared__ _Float16 sIn[4][64][48];   // halo tile: 64 chans x 47 samples (pad 48)
  int wv = threadIdx.x >> 5, lane = threadIdx.x & 31;
  int hs = lane >> 4, nl = lane & 15;
  int tile = blockIdx.x * 4 + wv;
  int b = tile / (TT / 16);
  int t0 = (tile % (TT / 16)) * 16;
  __builtin_prefetch(w + (size_t)nl * 2048, 0, 1);         // global_prefetch
  for (int idx = lane; idx < 64 * 47; idx += 32) {
    int hi = idx / 47, pos = idx % 47;
    int t = t0 + pos - 15;
    float vv = (t >= 0 && t < TT) ? in[((size_t)b * 64 + hi) * TT + t] : 0.f;
    sIn[wv][hi][pos] = (_Float16)vv;
  }
  __syncthreads();
  for (int n0 = 0; n0 < 64; n0 += 16) {
    v8f c;
    float bv = bias[n0 + nl];
#pragma unroll
    for (int r = 0; r < 8; ++r) c[r] = bv;
    for (int s = 0; s < 64; ++s) {      // K-step s == input channel hi
      v16h a, bt;
#pragma unroll
      for (int i = 0; i < 16; ++i) {
        a[i] = sIn[wv][s][nl + KMAPA(i, hs)];
        bt[i] = w[(size_t)(n0 + nl) * 2048 + s * 32 + KMAPB(i, hs)];
      }
      c = wmma32f16(a, bt, c);
    }
#pragma unroll
    for (int r = 0; r < 8; ++r) {
      int mm = r + (hs << 3);
      out[((size_t)b * TT + t0 + mm) * 64 + n0 + nl] = c[r];
    }
  }
}

// ---------------------------------------------------------------- timestep embedding MLP (tiny)
__global__ void __launch_bounds__(64) k_temb(const float* __restrict__ ts,
                                             const float* __restrict__ w1, const float* __restrict__ b1,
                                             const float* __restrict__ w2, const float* __restrict__ b2,
                                             float* __restrict__ cond, float* __restrict__ scb) {
  __shared__ float e[256];
  __shared__ float h1[64];
  int b = blockIdx.x, h = threadIdx.x;
  float t = ts[b];
  for (int j = h; j < 256; j += 64) {
    int f = (j < 128) ? j : (j - 128);
    float fr = expf(-logf(10000.f) * (float)f / 128.f);
    float arg = t * fr;
    e[j] = (j < 128) ? cosf(arg) : sinf(arg);
  }
  __syncthreads();
  float s = b1[h];
  for (int j = 0; j < 256; ++j) s += e[j] * w1[h * 256 + j];
  h1[h] = s / (1.f + expf(-s));
  __syncthreads();
  float c = b2[h];
  for (int j = 0; j < 64; ++j) c += h1[j] * w2[h * 64 + j];
  cond[b * 64 + h] = c;
  scb[b * 64 + h] = c / (1.f + expf(-c));   // sc = silu(cond)
}

// ---------------------------------------------------------------- adaLN projections (tiny)
__global__ void __launch_bounds__(256) k_ada(const float* __restrict__ scb,
                                             const float* __restrict__ aw, const float* __restrict__ ab,
                                             float* __restrict__ mall) {
  int blk = blockIdx.x;
  int i = blk / BB, b = blk % BB;
  for (int j = threadIdx.x; j < 768; j += 256) {
    float s = ab[i * 768 + j];
    for (int h = 0; h < 64; ++h) s += scb[b * 64 + h] * aw[((size_t)i * 768 + j) * 64 + h];
    mall[((size_t)i * BB + b) * 768 + j] = s;
  }
}

__global__ void __launch_bounds__(128) k_fm(const float* __restrict__ scb,
                                            const float* __restrict__ fw, const float* __restrict__ fb,
                                            float* __restrict__ fm) {
  int b = blockIdx.x, j = threadIdx.x;
  float s = fb[j];
  for (int h = 0; h < 64; ++h) s += scb[b * 64 + h] * fw[j * 64 + h];
  fm[b * 128 + j] = s;
}

// ---------------------------------------------------------------- LN + adaLN modulate (per token)
__global__ void __launch_bounds__(64) k_lnmod(const float* __restrict__ X,
                                              const float* __restrict__ mvec, int jsh, int jsc,
                                              float* __restrict__ out) {
  __shared__ float r1[64], r2[64];
  __shared__ float mv[2];
  int tok = blockIdx.x, h = threadIdx.x;
  int b = tok / TT;
  float x = X[(size_t)tok * 64 + h];
  r1[h] = x; r2[h] = x * x;
  __syncthreads();
  for (int s = 32; s > 0; s >>= 1) { if (h < s) { r1[h] += r1[h + s]; r2[h] += r2[h + s]; } __syncthreads(); }
  if (h == 0) { float m = r1[0] * (1.f / 64.f); mv[0] = m; mv[1] = rsqrtf(r2[0] * (1.f / 64.f) - m * m + 1e-6f); }
  __syncthreads();
  float shv = mvec[b * 768 + jsh * 64 + h];
  float scv = mvec[b * 768 + jsc * 64 + h];
  out[(size_t)tok * 64 + h] = ((x - mv[0]) * mv[1]) * (1.f + scv) + shv;
}

// ---------------------------------------------------------------- S5: Bu = u @ Bbar.T  (WMMA, N=P=32, K=H=64) -> (B,P,T)
__global__ void __launch_bounds__(128) k_bu_wmma(const float* __restrict__ u,
                                                 const _Float16* __restrict__ Bh, // (P,H) f16, coef folded
                                                 float* __restrict__ Bu) {
  __shared__ _Float16 sA[4][16][64];
  int wv = threadIdx.x >> 5, lane = threadIdx.x & 31;
  int hs = lane >> 4, nl = lane & 15;
  int tile = blockIdx.x * 4 + wv;
  int tok0 = tile * 16;
  int b = tok0 / TT;
  int t0 = tok0 - b * TT;
  for (int idx = lane; idx < 16 * 64; idx += 32) {
    int mm = idx >> 6, hh = idx & 63;
    sA[wv][mm][hh] = (_Float16)u[(size_t)(tok0 + mm) * 64 + hh];
  }
  __syncthreads();
  for (int n0 = 0; n0 < 32; n0 += 16) {
    v8f c;
#pragma unroll
    for (int r = 0; r < 8; ++r) c[r] = 0.f;
#pragma unroll
    for (int ks = 0; ks < 64; ks += 32) {
      v16h a, bt;
#pragma unroll
      for (int i = 0; i < 16; ++i) {
        a[i] = sA[wv][nl][ks + KMAPA(i, hs)];
        bt[i] = Bh[(size_t)(n0 + nl) * 64 + ks + KMAPB(i, hs)];
      }
      c = wmma32f16(a, bt, c);
    }
#pragma unroll
    for (int r = 0; r < 8; ++r) {
      int mm = r + (hs << 3);
      Bu[((size_t)b * PP + n0 + nl) * TT + t0 + mm] = c[r];
    }
  }
}

// ---------------------------------------------------------------- S5 fwd+bwd blocked scan -> xcat (B,T,2P)
__global__ void __launch_bounds__(128) k_scan(const float* __restrict__ Bu,
                                              const float* __restrict__ lam_log,
                                              const float* __restrict__ log_dt,
                                              float* __restrict__ xcat) {
  constexpr int CH = TT / 128;  // 32 per thread
  int bp = blockIdx.x;
  int b = bp / PP, pp = bp % PP;
  float lam = -expf(lam_log[pp]);
  float A = expf(lam * expf(log_dt[pp]));
  const float* bu = Bu + (size_t)bp * TT;
  __shared__ float cl[128], pf[128];
  int j = threadIdx.x;
  float v[CH];
  // forward: x[t] = A x[t-1] + u[t]
  { int s = j * CH; float x = 0.f;
    for (int t = 0; t < CH; ++t) { x = A * x + bu[s + t]; v[t] = x; }
    cl[j] = x; }
  __syncthreads();
  if (j == 0) {
    float AL = 1.f; for (int k = 0; k < CH; ++k) AL *= A;
    float X = 0.f; for (int q = 0; q < 128; ++q) { pf[q] = X; X = AL * X + cl[q]; }
  }
  __syncthreads();
  { int s = j * CH; float P0 = pf[j]; float mlt = A;
    for (int t = 0; t < CH; ++t) { xcat[((size_t)b * TT + s + t) * 64 + pp] = v[t] + mlt * P0; mlt *= A; } }
  __syncthreads();
  // backward (flip)
  { int e = TT - 1 - j * CH; float x = 0.f;
    for (int t = 0; t < CH; ++t) { x = A * x + bu[e - t]; v[t] = x; }
    cl[j] = x; }
  __syncthreads();
  if (j == 0) {
    float AL = 1.f; for (int k = 0; k < CH; ++k) AL *= A;
    float X = 0.f; for (int q = 0; q < 128; ++q) { pf[q] = X; X = AL * X + cl[q]; }
  }
  __syncthreads();
  { int e = TT - 1 - j * CH; float P0 = pf[j]; float mlt = A;
    for (int t = 0; t < CH; ++t) { xcat[((size_t)b * TT + (e - t)) * 64 + 32 + pp] = v[t] + mlt * P0; mlt *= A; } }
}

// ---------------------------------------------------------------- S5 output projection (WMMA) + skip + gate
// out[tok,h] = g[b,h] * ( xcat[tok,:] @ C[h,:] + u[tok,h]*D[h] )
__global__ void __launch_bounds__(128) k_s5out_wmma(const float* __restrict__ xcat,
                                                    const float* __restrict__ u,
                                                    const _Float16* __restrict__ Ch, // (H, 2P) f16
                                                    const float* __restrict__ D,
                                                    const float* __restrict__ mvec, int jg,
                                                    float* __restrict__ out) {
  __shared__ _Float16 sA[4][16][64];
  int wv = threadIdx.x >> 5, lane = threadIdx.x & 31;
  int hs = lane >> 4, nl = lane & 15;
  int tile = blockIdx.x * 4 + wv;
  int tok0 = tile * 16;
  int b = tok0 / TT;
  for (int idx = lane; idx < 16 * 64; idx += 32) {
    int mm = idx >> 6, hh = idx & 63;
    sA[wv][mm][hh] = (_Float16)xcat[(size_t)(tok0 + mm) * 64 + hh];
  }
  __syncthreads();
  for (int n0 = 0; n0 < 64; n0 += 16) {
    v8f c;
#pragma unroll
    for (int r = 0; r < 8; ++r) c[r] = 0.f;
#pragma unroll
    for (int ks = 0; ks < 64; ks += 32) {
      v16h a, bt;
#pragma unroll
      for (int i = 0; i < 16; ++i) {
        a[i] = sA[wv][nl][ks + KMAPA(i, hs)];
        bt[i] = Ch[(size_t)(n0 + nl) * 64 + ks + KMAPB(i, hs)];
      }
      c = wmma32f16(a, bt, c);
    }
    int col = n0 + nl;
    float Dv = D[col];
    float g = mvec[b * 768 + jg * 64 + col];
#pragma unroll
    for (int r = 0; r < 8; ++r) {
      int mm = r + (hs << 3);
      size_t idx = (size_t)(tok0 + mm) * 64 + col;
      out[idx] = g * (c[r] + u[idx] * Dv);
    }
  }
}

// ---------------------------------------------------------------- fused adaLN-MLP: (opt LN+mod) -> GEMM1 -> GELU -> GEMM2 -> epilogue, WMMA
// emode: 0: out=y   1: out=res+y   2: out=res+g*y   3: out+=res+g*y
__global__ void __launch_bounds__(128) k_mlp(const float* __restrict__ X,
                                             const _Float16* __restrict__ w1, const float* __restrict__ b1,
                                             const _Float16* __restrict__ w2, const float* __restrict__ b2,
                                             const float* __restrict__ mvec, int jsh, int jsc, int jg,
                                             int lnflag, const float* __restrict__ res,
                                             float* __restrict__ out, int emode) {
  __shared__ float    stage[4][16][65];    // 16 tokens x 64 features (f32, for LN)
  __shared__ _Float16 sA[4][16][64];       // f16 A-operand tile
  __shared__ _Float16 hid[4][16][264];     // 16 x 256 hidden (f16), never leaves LDS
  int wv = threadIdx.x >> 5, lane = threadIdx.x & 31;
  int hs = lane >> 4, nl = lane & 15;
  int tile = blockIdx.x * 4 + wv;
  int tok0 = tile * 16;
  int b = tok0 / TT;
  __builtin_prefetch(w2 + (size_t)nl * 256, 0, 1);   // global_prefetch of GEMM2 weights
  for (int idx = lane; idx < 16 * 64; idx += 32) {
    int mm = idx >> 6, hh = idx & 63;
    stage[wv][mm][hh] = X[(size_t)(tok0 + mm) * 64 + hh];
  }
  __syncthreads();
  if (lnflag) {
    if (lane < 16) {
      float s = 0.f, s2 = 0.f;
      for (int hh = 0; hh < 64; ++hh) { float xv = stage[wv][lane][hh]; s += xv; s2 += xv * xv; }
      float mean = s * (1.f / 64.f);
      float rs = rsqrtf(s2 * (1.f / 64.f) - mean * mean + 1e-6f);
      for (int hh = 0; hh < 64; ++hh) {
        float shv = mvec[b * 768 + jsh * 64 + hh];
        float scv = mvec[b * 768 + jsc * 64 + hh];
        float xv = (stage[wv][lane][hh] - mean) * rs;
        stage[wv][lane][hh] = xv * (1.f + scv) + shv;
      }
    }
  }
  __syncthreads();
  for (int idx = lane; idx < 16 * 64; idx += 32) {     // f32 -> f16 A tile
    int mm = idx >> 6, hh = idx & 63;
    sA[wv][mm][hh] = (_Float16)stage[wv][mm][hh];
  }
  __syncthreads();
  // GEMM1: (16x64) @ (64x256) + b1, GELU, keep in LDS as f16
  for (int n0 = 0; n0 < 256; n0 += 16) {
    v8f c;
    float bias = b1[n0 + nl];
#pragma unroll
    for (int r = 0; r < 8; ++r) c[r] = bias;
#pragma unroll
    for (int ks = 0; ks < 64; ks += 32) {
      v16h a, bt;
#pragma unroll
      for (int i = 0; i < 16; ++i) {
        a[i] = sA[wv][nl][ks + KMAPA(i, hs)];
        bt[i] = w1[(size_t)(n0 + nl) * 64 + ks + KMAPB(i, hs)];  // w1.T (K=H, N=MH)
      }
      c = wmma32f16(a, bt, c);
    }
#pragma unroll
    for (int r = 0; r < 8; ++r) {
      int mm = r + (hs << 3);
      float xg = c[r];
      float g = 0.5f * xg * (1.f + erff(xg * 0.70710678f));      // exact GELU
      hid[wv][mm][n0 + nl] = (_Float16)g;
    }
  }
  __syncthreads();
  // GEMM2: (16x256) @ (256x64) + b2, epilogue
  for (int n0 = 0; n0 < 64; n0 += 16) {
    v8f c;
    float bias = b2[n0 + nl];
#pragma unroll
    for (int r = 0; r < 8; ++r) c[r] = bias;
    for (int ks = 0; ks < 256; ks += 32) {
      v16h a, bt;
#pragma unroll
      for (int i = 0; i < 16; ++i) {
        a[i] = hid[wv][nl][ks + KMAPA(i, hs)];
        bt[i] = w2[(size_t)(n0 + nl) * 256 + ks + KMAPB(i, hs)]; // w2.T (K=MH, N=H)
      }
      c = wmma32f16(a, bt, c);
    }
#pragma unroll
    for (int r = 0; r < 8; ++r) {
      int mm = r + (hs << 3);
      size_t idx = (size_t)(tok0 + mm) * 64 + n0 + nl;
      float y = c[r];
      float o;
      if (emode == 0)      o = y;
      else if (emode == 1) o = res[idx] + y;
      else                 o = res[idx] + mvec[b * 768 + jg * 64 + n0 + nl] * y;
      if (emode == 3) out[idx] += o; else out[idx] = o;
    }
  }
}

// ---------------------------------------------------------------- final LN + mod + projection + flow update
__global__ void __launch_bounds__(64) k_final(const float* __restrict__ acc,
                                              const float* __restrict__ fm,
                                              const float* __restrict__ flw,
                                              const float* __restrict__ flb,
                                              const float* __restrict__ xt,
                                              const float* __restrict__ ts,
                                              float* __restrict__ out) {
  __shared__ float r1[64], r2[64];
  __shared__ float mv[2];
  int tok = blockIdx.x, h = threadIdx.x;
  int b = tok / TT;
  float x = acc[(size_t)tok * 64 + h];
  r1[h] = x; r2[h] = x * x;
  __syncthreads();
  for (int s = 32; s > 0; s >>= 1) { if (h < s) { r1[h] += r1[h + s]; r2[h] += r2[h + s]; } __syncthreads(); }
  if (h == 0) { float m = r1[0] * (1.f / 64.f); mv[0] = m; mv[1] = rsqrtf(r2[0] * (1.f / 64.f) - m * m + 1e-6f); }
  __syncthreads();
  float modv = ((x - mv[0]) * mv[1]) * (1.f + fm[b * 128 + 64 + h]) + fm[b * 128 + h];
  r1[h] = modv * flw[h];
  __syncthreads();
  for (int s = 32; s > 0; s >>= 1) { if (h < s) r1[h] += r1[h + s]; __syncthreads(); }
  if (h == 0) out[tok] = xt[tok] + (1.f - ts[b]) * (r1[0] + flb[0]);
}

// ================================================================ host
extern "C" void kernel_launch(void* const* d_in, const int* in_sizes, int n_in,
                              void* d_out, int out_size, void* d_ws, size_t ws_size,
                              hipStream_t stream) {
  (void)in_sizes; (void)n_in; (void)out_size; (void)ws_size;
  const float* ppg    = (const float*)d_in[0];
  const float* tgt    = (const float*)d_in[1];
  const float* tstep  = (const float*)d_in[2];
  const float* x0     = (const float*)d_in[3];
  const float* pc1_w  = (const float*)d_in[4];
  const float* pc1_b  = (const float*)d_in[5];
  const float* pc2_w  = (const float*)d_in[6];
  const float* pc2_b  = (const float*)d_in[7];
  const float* tc1_w  = (const float*)d_in[8];
  const float* tc1_b  = (const float*)d_in[9];
  const float* tc2_w  = (const float*)d_in[10];
  const float* tc2_b  = (const float*)d_in[11];
  const float* te_w1  = (const float*)d_in[12];
  const float* te_b1  = (const float*)d_in[13];
  const float* te_w2  = (const float*)d_in[14];
  const float* te_b2  = (const float*)d_in[15];
  const float* ada_w  = (const float*)d_in[16];
  const float* ada_b  = (const float*)d_in[17];
  const float* sp_lam = (const float*)d_in[18];
  const float* sp_B   = (const float*)d_in[19];
  const float* sp_C   = (const float*)d_in[20];
  const float* sp_D   = (const float*)d_in[21];
  const float* sp_dt  = (const float*)d_in[22];
  const float* st_lam = (const float*)d_in[23];
  const float* st_B   = (const float*)d_in[24];
  const float* st_C   = (const float*)d_in[25];
  const float* st_D   = (const float*)d_in[26];
  const float* st_dt  = (const float*)d_in[27];
  const float* mlp_w1 = (const float*)d_in[28];
  const float* mlp_b1 = (const float*)d_in[29];
  const float* mlp_w2 = (const float*)d_in[30];
  const float* mlp_b2 = (const float*)d_in[31];
  const float* fa_w   = (const float*)d_in[32];
  const float* fa_b   = (const float*)d_in[33];
  const float* fl_w   = (const float*)d_in[34];
  const float* fl_b   = (const float*)d_in[35];
  float* outp = (float*)d_out;

  float* ws = (float*)d_ws;
  size_t off = 0;
  auto alloc = [&](size_t n) { float* r = ws + off; off += n; return r; };
  float* xt    = alloc(NTOK);
  float* c1    = alloc(NH);            // conv1 output; reused as p_mid later
  float* p     = alloc(NH);
  float* xe    = alloc(NH);
  float* acc   = alloc(NH);
  float* u     = alloc(NH);
  float* xc    = alloc(NH);            // scan output / pa / t2in
  float* pcond = alloc(NH);
  float* tcond = alloc(NH);            // t_cond / xt_new
  float* bu    = alloc((size_t)NTOK * PP);
  float* cond  = alloc(BB * HH);
  float* scb   = alloc(BB * HH);
  float* mall  = alloc((size_t)NBLK_ * BB * 768);
  float* fmb   = alloc(BB * 2 * HH);
  float* pmid  = c1;
  // f16 pre-converted weights, carved after the f32 region
  _Float16* hb = (_Float16*)(ws + off);
  size_t hoff = 0;
  auto halloc = [&](size_t n) { _Float16* r = hb + hoff; hoff += n; return r; };
  _Float16* w1h  = halloc((size_t)NBLK_ * 5 * MHID * HH);  // 327680
  _Float16* w2h  = halloc((size_t)NBLK_ * 5 * HH * MHID);  // 327680
  _Float16* cwp  = halloc((size_t)HH * HH * 32);           // 131072
  _Float16* cwt  = halloc((size_t)HH * HH * 32);           // 131072
  _Float16* spCh = halloc((size_t)NBLK_ * HH * 2 * PP);    // 16384
  _Float16* stCh = halloc((size_t)NBLK_ * HH * 2 * PP);    // 16384
  _Float16* spBh = halloc((size_t)NBLK_ * PP * HH);        // 8192
  _Float16* stBh = halloc((size_t)NBLK_ * PP * HH);        // 8192

  const int GNH = (int)(NH / 256);        // 32768
  const int GTK = NTOK;                   // token-parallel grids
  const int GMM = NTOK / 16 / 4;          // 2048: 16-token tiles, 4 waves/block

  // weight prep (f32 -> f16, coef folding)
  k_cvt<<<327680 / 256, 256, 0, stream>>>(mlp_w1, w1h);
  k_cvt<<<327680 / 256, 256, 0, stream>>>(mlp_w2, w2h);
  k_cvt<<<131072 / 256, 256, 0, stream>>>(pc2_w, cwp);
  k_cvt<<<131072 / 256, 256, 0, stream>>>(tc2_w, cwt);
  k_cvt<<<16384 / 256, 256, 0, stream>>>(sp_C, spCh);
  k_cvt<<<16384 / 256, 256, 0, stream>>>(st_C, stCh);
  k_bbar<<<(NBLK_ * PP * HH) / 256, 256, 0, stream>>>(sp_lam, sp_B, sp_dt, spBh);
  k_bbar<<<(NBLK_ * PP * HH) / 256, 256, 0, stream>>>(st_lam, st_B, st_dt, stBh);

  k_xt<<<NTOK / 256, 256, 0, stream>>>(tgt, x0, tstep, xt);
  k_conv1<<<GNH, 256, 0, stream>>>(ppg, pc1_w, pc1_b, c1);
  k_conv2<<<GMM, 128, 0, stream>>>(c1, cwp, pc2_b, p);
  k_conv1<<<GNH, 256, 0, stream>>>(xt, tc1_w, tc1_b, c1);
  k_conv2<<<GMM, 128, 0, stream>>>(c1, cwt, tc2_b, xe);
  k_temb<<<BB, 64, 0, stream>>>(tstep, te_w1, te_b1, te_w2, te_b2, cond, scb);
  k_ada<<<NBLK_ * BB, 256, 0, stream>>>(scb, ada_w, ada_b, mall);
  k_fm<<<BB, 128, 0, stream>>>(scb, fa_w, fa_b, fmb);
  k_zero<<<GNH, 256, 0, stream>>>(acc);

  for (int i = 0; i < NBLK_; ++i) {
    const float* mi = mall + (size_t)i * BB * 768;
    const _Float16* W1_ = nullptr; const float* B1_ = nullptr;
    const _Float16* W2_ = nullptr; const float* B2_ = nullptr;
    auto setw = [&](int s) {
      W1_ = w1h + (size_t)(i * 5 + s) * MHID * HH;
      B1_ = mlp_b1 + (size_t)(i * 5 + s) * MHID;
      W2_ = w2h + (size_t)(i * 5 + s) * HH * MHID;
      B2_ = mlp_b2 + (size_t)(i * 5 + s) * HH;
    };
    // ---- p path: S5
    k_lnmod<<<GTK, 64, 0, stream>>>(p, mi, 0, 1, u);
    k_bu_wmma<<<GMM, 128, 0, stream>>>(u, spBh + (size_t)i * PP * HH, bu);
    k_scan<<<BB * PP, 128, 0, stream>>>(bu, sp_lam + i * PP, sp_dt + i * PP, xc);
    k_s5out_wmma<<<GMM, 128, 0, stream>>>(xc, u, spCh + (size_t)i * HH * 2 * PP, sp_D + i * HH, mi, 2, pcond);
    k_add<<<GNH, 256, 0, stream>>>(p, pcond, pmid);                              // p_mid = p + p_cond
    setw(1);
    k_mlp<<<GMM, 128, 0, stream>>>(pmid, W1_, B1_, W2_, B2_, mi, 3, 4, 5, 1, p, p, 2);  // p = p + g_mp*MLP(lnmod(p_mid))
    // ---- t path: S5
    k_lnmod<<<GTK, 64, 0, stream>>>(xe, mi, 6, 7, u);
    k_bu_wmma<<<GMM, 128, 0, stream>>>(u, stBh + (size_t)i * PP * HH, bu);
    k_scan<<<BB * PP, 128, 0, stream>>>(bu, st_lam + i * PP, st_dt + i * PP, xc);
    k_s5out_wmma<<<GMM, 128, 0, stream>>>(xc, u, stCh + (size_t)i * HH * 2 * PP, st_D + i * HH, mi, 8, tcond);
    // ---- cross MLPs
    setw(0);
    k_mlp<<<GMM, 128, 0, stream>>>(pcond, W1_, B1_, W2_, B2_, mi, 0, 0, 0, 0, nullptr, xc, 0);   // pa -> xc
    setw(2);
    k_mlp<<<GMM, 128, 0, stream>>>(tcond, W1_, B1_, W2_, B2_, mi, 0, 0, 0, 0, nullptr, u, 0);    // ta -> u
    k_add<<<GNH, 256, 0, stream>>>(xc, u, xc);                                   // ta + pa
    setw(3);
    k_mlp<<<GMM, 128, 0, stream>>>(xc, W1_, B1_, W2_, B2_, mi, 0, 0, 0, 0, xe, tcond, 1);        // xt_new = xe + MLP(ta+pa)
    setw(4);
    k_mlp<<<GMM, 128, 0, stream>>>(tcond, W1_, B1_, W2_, B2_, mi, 9, 10, 11, 1, xe, acc, 3);     // acc += xe + g_mt*MLP(lnmod(xt_new))
  }
  k_final<<<GTK, 64, 0, stream>>>(acc, fmb, fl_w, fl_b, xt, tstep, outp);
}
